// Simulation_37864431681582
// MI455X (gfx1250) — matrix operand order repaired and checked by reading, not compiled, not run on hardware
//
#include <hip/hip_runtime.h>
#include <stdint.h>
#include <math.h>

typedef __attribute__((ext_vector_type(16))) _Float16 v16h;
typedef __attribute__((ext_vector_type(8)))  float    v8f;

#define GRID_N 128
#define NE     4096
#define SIGMA  1.5f

// ---------- deterministic counter-based RNG (per-pixel stream) ----------
__device__ __forceinline__ uint32_t hash32(uint32_t x) {
    x ^= x >> 17; x *= 0xed5ad4bbu;
    x ^= x >> 11; x *= 0xac4c1b51u;
    x ^= x >> 15; x *= 0x31848babu;
    x ^= x >> 14;
    return x;
}

struct RNG {
    uint32_t s;
    __device__ __forceinline__ float next() {
        s = s * 747796405u + 2891336453u;
        uint32_t w = hash32(s);
        return (float)(w >> 8) * (1.0f / 16777216.0f) + (0.5f / 16777216.0f); // (0,1)
    }
};

__device__ __forceinline__ float normal_sample(RNG& r) {
    float u1 = r.next();
    float u2 = r.next();
    float rad = sqrtf(-2.0f * __logf(u1));
    return rad * __cosf(6.28318530717958f * u2);
}

__device__ __forceinline__ float poisson_sample(RNG& r, float lam) {
    if (lam < 12.0f) {
        // Knuth multiplication method (bounded for determinism/safety)
        float L = __expf(-lam);
        float p = 1.0f;
        float k = -1.0f;
        #pragma unroll 1
        for (int it = 0; it < 256; ++it) {
            k += 1.0f;
            p *= r.next();
            if (p <= L) break;
        }
        return k;
    }
    // normal approximation for large lambda
    float v = floorf(lam + sqrtf(lam) * normal_sample(r) + 0.5f);
    return v < 0.0f ? 0.0f : v;
}

__device__ __forceinline__ float gamma_sample(RNG& r, float alpha) {
    float boost = 1.0f;
    if (alpha < 1.0f) {
        // Gamma(a) = Gamma(a+1) * U^(1/a)
        float u = r.next();
        boost = __expf(__logf(u) / alpha);   // underflows to 0 for tiny alpha (correct regime)
        alpha += 1.0f;
    }
    // Marsaglia-Tsang squeeze/rejection (bounded)
    float d = alpha - (1.0f / 3.0f);
    float c = 1.0f / sqrtf(9.0f * d);
    #pragma unroll 1
    for (int it = 0; it < 64; ++it) {
        float x = normal_sample(r);
        float v = 1.0f + c * x;
        if (v <= 0.0f) continue;
        v = v * v * v;
        float u  = r.next();
        float x2 = x * x;
        if (u < 1.0f - 0.0331f * x2 * x2) return d * v * boost;
        if (__logf(u) < 0.5f * x2 + d * (1.0f - v + __logf(v))) return d * v * boost;
    }
    return d * boost;
}

// A-fragment k index for VGPR j, half h, lane-half hi (ISA 16-bit A 16x32 layout):
// VGPRs 0-3: K=0-7 (lanes 0-15) / K=8-15 (lanes 16-31); VGPRs 4-7: K=16-23 / K=24-31.
__device__ __forceinline__ int a_kidx(int j, int h, int hi) {
    return (j < 4) ? (j * 2 + h + hi * 8) : (16 + (j - 4) * 2 + h + hi * 8);
}

// ---------- fused separable-PSF GEMM (WMMA) + camera-noise sampling ----------
// pdf = A(128x4096) * B(4096x128) with A[i,e]=exp(-(i-muy_e)^2/2s^2),
// B[e,j]=I_e*norm*exp(-(j-mux_e)^2/2s^2). Block = one 16-row tile band (mt);
// its 8 waves cover the 8 column tiles (nt). A is block-invariant, so it is
// precomputed once into LDS in the exact WMMA A-register image and re-fetched
// with ds_load_b128 by every wave.
__global__ void __launch_bounds__(256)
psf_sim_kernel(const float* __restrict__ positions,  // [4096][3] = x, y, I
               const float* __restrict__ bg,         // [128][128]
               const float* __restrict__ qe_p,
               const float* __restrict__ dark_p,
               const float* __restrict__ expo_p,
               const float* __restrict__ gain_p,
               const float* __restrict__ ro_p,
               float* __restrict__ out)              // [128][128]
{
    __shared__ float sx[NE];        // emitter x
    __shared__ float sy[NE];        // emitter y
    __shared__ float si[NE];        // I * norm (pre-scaled)
    // A panel: 128 k-blocks x 32 lanes, each slot is one lane's v16h register
    // image (32 B). 128 KB — only viable with CDNA5's 320 KB/WGP LDS.
    __shared__ v16h A_panel[(NE / 32) * 32];

    const float inv_var = 1.0f / (SIGMA * SIGMA);
    const float nhalf   = -0.5f * inv_var;
    const float norm    = 0.07073553f;       // 1/(2*pi*sigma^2), sigma=1.5

    const int tid = threadIdx.x;
    const int mt  = blockIdx.x;              // output tile row (i / 16)

    // Stage all emitters into LDS once (48 KB).
    for (int e = tid; e < NE; e += 256) {
        float x = positions[3 * e + 0];
        float y = positions[3 * e + 1];
        float I = positions[3 * e + 2];
        sx[e] = x;
        sy[e] = y;
        si[e] = I * norm;
    }
    __syncthreads();

    // Cooperatively build the block-invariant A panel (16 rows x 4096 emitters,
    // f16, WMMA register layout). 4096 slots / 256 threads = 16 slots each.
    for (int s = tid; s < (NE / 32) * 32; s += 256) {
        const int kb     = s >> 5;           // k-block (32 emitters)
        const int lane_s = s & 31;
        const int sub_s  = lane_s & 15;
        const int hi_s   = lane_s >> 4;
        const float fm   = (float)(mt * 16 + sub_s);   // image row i
        v16h av;
        #pragma unroll
        for (int j = 0; j < 8; ++j) {
            #pragma unroll
            for (int h = 0; h < 2; ++h) {
                int k = kb * 32 + a_kidx(j, h, hi_s);
                float dy = fm - sy[k];
                av[2 * j + h] = (_Float16)__expf(nhalf * dy * dy);
            }
        }
        A_panel[s] = av;
    }
    __syncthreads();

    const int lane = tid & 31;
    const int sub  = lane & 15;   // col-within-tile (B, C)
    const int hi   = lane >> 4;   // lane half selector
    const int nt   = tid >> 5;    // output tile col (j / 16), one per wave

    const float fj = (float)(nt * 16 + sub);  // image col j -> pairs with emitter x

    v8f c = {};

    // K-loop: a comes straight from LDS (2x ds_load_b128); only B's 16 exps
    // remain inline. EXEC uniform throughout — required for v_wmma.
    for (int kb = 0; kb < NE / 32; ++kb) {
        v16h a = A_panel[(kb << 5) + lane];
        v16h b;
        #pragma unroll
        for (int j = 0; j < 8; ++j) {
            #pragma unroll
            for (int h = 0; h < 2; ++h) {
                // B 32x16 f16 layout: same k order as A (K=0-7,16-23 lanes 0-15; K=8-15,24-31 lanes 16-31)
                int kB = (kb << 5) + a_kidx(j, h, hi);
                float dx = fj - sx[kB];
                b[2 * j + h] = (_Float16)(si[kB] * __expf(nhalf * dx * dx));
            }
        }
        c = __builtin_amdgcn_wmma_f32_16x16x32_f16(
                /*neg_a=*/false, a, /*neg_b=*/false, b,
                /*c_mod=*/(short)0, c, /*reuse_a=*/false, /*reuse_b=*/false);
    }

    // Epilogue: lambda -> Poisson -> Gamma -> + Gaussian readout noise.
    const float QE = qe_p[0];
    const float DK = dark_p[0] * expo_p[0];
    const float G  = gain_p[0];
    const float RO = ro_p[0];

    // C/D f32 16x16 layout: VGPR r holds M = r + 8*hi, lane sub = N.
    #pragma unroll 1
    for (int r = 0; r < 8; ++r) {
        int row = mt * 16 + hi * 8 + r;
        int col = nt * 16 + sub;
        int idx = row * GRID_N + col;

        float lam = c[r] * QE + DK + floorf(bg[idx] * 0.5f) + 1e-5f;

        RNG rng;
        rng.s = hash32(0x9E3779B9u ^ (uint32_t)idx);

        float pois = poisson_sample(rng, lam);
        float gam  = gamma_sample(rng, pois + 1e-5f) * G;
        out[idx]   = gam + RO * normal_sample(rng);
    }
}

extern "C" void kernel_launch(void* const* d_in, const int* in_sizes, int n_in,
                              void* d_out, int out_size, void* d_ws, size_t ws_size,
                              hipStream_t stream) {
    const float* positions = (const float*)d_in[0];   // 4096*3
    const float* bg        = (const float*)d_in[1];   // 128*128
    const float* qe        = (const float*)d_in[2];
    const float* dark      = (const float*)d_in[3];
    const float* expo      = (const float*)d_in[4];
    const float* gain      = (const float*)d_in[5];
    const float* ro        = (const float*)d_in[6];
    float* out             = (float*)d_out;           // 128*128

    dim3 grid(GRID_N / 16);   // 8 blocks: one 16-row tile band each
    dim3 block(256);          // 8 wave32 waves: one 16x16 output tile each
    hipLaunchKernelGGL(psf_sim_kernel, grid, block, 0, stream,
                       positions, bg, qe, dark, expo, gain, ro, out);
}
